// Bottleneck_dcn_10196252361291
// MI455X (gfx1250) — matrix operand order repaired and checked by reading, not compiled, not run on hardware
//
#include <hip/hip_runtime.h>
#include <hip/hip_bf16.h>

typedef __attribute__((ext_vector_type(8)))  __bf16 v8bf;
typedef __attribute__((ext_vector_type(16))) __bf16 v16bf;
typedef __attribute__((ext_vector_type(8)))  float  v8f;

#define MBLK 256
#define NBLK 64
#define KSTEP 32
#define LDSS 40   // LDS row stride in bf16 elements (80B, 16B aligned, conflict-padded)

__device__ __forceinline__ float silu_fast(float o) {
    // o * sigmoid(o) with fast v_rcp_f32 (bf16-precision pipeline)
    return o * __builtin_amdgcn_rcpf(1.0f + __expf(-o));
}

// ---------------------------------------------------------------------------
// Prep kernels
// ---------------------------------------------------------------------------

// NCHW f32 -> NHWC bf16  (4,256,64,64) -> [16384][256]
__global__ void pack_x_k(const float* __restrict__ x, __bf16* __restrict__ xb) {
    int idx = blockIdx.x * 256 + threadIdx.x;          // over 4*256*4096
    int sp  = idx & 4095;
    int c   = (idx >> 12) & 255;
    int n   = idx >> 20;
    xb[((size_t)(n * 4096 + sp)) * 256 + c] = (__bf16)x[idx];
}

// OIHW f32 -> Bt bf16 [Opad][T*I], k = tap*I + cin; rows >= O zero-filled
__global__ void prep_w_k(const float* __restrict__ src, __bf16* __restrict__ dst,
                         int O, int I, int T, int Opad) {
    int idx = blockIdx.x * 256 + threadIdx.x;
    int ti  = T * I;
    long total = (long)Opad * ti;
    if (idx >= total) return;
    int o   = idx / ti;
    int r   = idx - o * ti;
    int tap = r / I;
    int c   = r - tap * I;
    dst[idx] = (o < O) ? (__bf16)src[(size_t)(o * I + c) * T + tap] : (__bf16)0.0f;
}

__global__ void bn_affine_k(const float* g, const float* b, const float* m, const float* v,
                            float* s, float* t, int n) {
    int i = blockIdx.x * 256 + threadIdx.x;
    if (i >= n) return;
    float sc = g[i] * rsqrtf(v[i] + 1e-5f);
    s[i] = sc;
    t[i] = b[i] - m[i] * sc;
}

__global__ void bias_affine_k(const float* bias, float* s, float* t, int n, int npad) {
    int i = blockIdx.x * 256 + threadIdx.x;
    if (i >= npad) return;
    s[i] = (i < n) ? 1.0f : 0.0f;
    t[i] = (i < n) ? bias[i] : 0.0f;
}

// fold outp_b into bn3:  (acc + outp_b)*s3 + t3 = acc*s3 + (outp_b*s3 + t3)
__global__ void outp_bn_affine_k(const float* ob, const float* g, const float* b,
                                 const float* m, const float* v, float* s, float* t, int n) {
    int i = blockIdx.x * 256 + threadIdx.x;
    if (i >= n) return;
    float s3 = g[i] * rsqrtf(v[i] + 1e-5f);
    float t3 = b[i] - m[i] * s3;
    s[i] = s3;
    t[i] = ob[i] * s3 + t3;
}

// ---------------------------------------------------------------------------
// Unified implicit-GEMM WMMA kernel (bf16 in, f32 accumulate)
//   256x64 block tile, 8 waves, each wave 32x64 (2x4 wmma tiles).
//   Register-staged LDS double buffer: next K-step global loads are issued
//   before the WMMA block; waits + ds_store land after it (latency hidden).
//   3x3 conv -> ntaps=9 with incremental tap/channel advance (no division).
//   epilogue: o = acc*scale[n] + shift[n]; optional SiLU
//   mode 0: f32 store   mode 1: bf16 store
//   mode 2: final residual -> NCHW f32: out = o + x_nchw + y2_nhwc
// ---------------------------------------------------------------------------
__global__ __launch_bounds__(256) void gemm_wmma_k(
    const __bf16* __restrict__ A, const __bf16* __restrict__ Bt,
    const float* __restrict__ scale, const float* __restrict__ shift,
    int K_total, int Cin, int N, int ldC, int ntaps, int do_silu, int mode,
    float* __restrict__ Cf, __bf16* __restrict__ Cb,
    const float* __restrict__ xres, const float* __restrict__ y2res,
    float* __restrict__ outf)
{
    __shared__ __align__(16) __bf16 Als[2][MBLK * LDSS];
    __shared__ __align__(16) __bf16 Bls[2][NBLK * LDSS];

    const int tid  = threadIdx.x;
    const int lane = tid & 31;
    const int wave = tid >> 5;
    const int mBase = blockIdx.y * MBLK;
    const int nBase = blockIdx.x * NBLK;

    v8f acc[2][4];
#pragma unroll
    for (int s = 0; s < 2; ++s)
#pragma unroll
        for (int j = 0; j < 4; ++j)
#pragma unroll
            for (int e = 0; e < 8; ++e) acc[s][j][e] = 0.0f;

    // staging roles: A -> thread t owns full row t (64B = 4x uint4)
    //                B -> row = t>>2, quarter = t&3 (16B)
    const int arow = tid;
    const int brow = tid >> 2, bq = tid & 3;
    const int gp = mBase + arow;
    const int pn = gp >> 12, ph = (gp >> 6) & 63, pw = gp & 63;

    const int hi = lane >> 4, ml = lane & 15;
    const int ksteps = K_total / KSTEP;
    const int kpc = Cin / KSTEP;   // K-steps per conv tap

    uint4 ra[4], rb;

    auto loadA = [&](int tap, int c0) {
        int src = gp;
        bool valid = true;
        if (ntaps > 1) {
            int t3 = tap / 3;                 // once per stage, tap in [0,8]
            int dy = t3 - 1, dx = tap - t3 * 3 - 1;
            int sy = ph + dy, sx = pw + dx;
            valid = (sy >= 0) & (sy < 64) & (sx >= 0) & (sx < 64);
            src = (pn << 12) + (sy << 6) + sx;
        }
        if (valid) {
            const uint4* p = (const uint4*)(A + (size_t)src * Cin + c0);
#pragma unroll
            for (int i = 0; i < 4; ++i) ra[i] = p[i];
        } else {
#pragma unroll
            for (int i = 0; i < 4; ++i) ra[i] = make_uint4(0, 0, 0, 0);
        }
    };
    auto loadB = [&](int k0) {
        rb = *(const uint4*)(Bt + (size_t)(nBase + brow) * K_total + k0 + bq * 8);
    };
    auto stage = [&](int buf) {
        uint4* da = (uint4*)&Als[buf][arow * LDSS];
#pragma unroll
        for (int i = 0; i < 4; ++i) da[i] = ra[i];
        *(uint4*)&Bls[buf][brow * LDSS + bq * 8] = rb;
    };

    // prologue: stage K-step 0 into buffer 0
    loadA(0, 0);
    loadB(0);
    stage(0);
    __syncthreads();

    int cur = 0;
    int tap_c = 0, kc_c = 0, k0_c = 0;
    for (int ks = 0; ks < ksteps; ++ks) {
        // advance pipeline indices (no division)
        int tap_n = tap_c, kc_n = kc_c + 1;
        if (kc_n == kpc) { kc_n = 0; ++tap_n; }
        const bool has_next = (ks + 1 < ksteps);
        if (has_next) {                       // issue next global loads early
            loadA(tap_n, kc_n * KSTEP);
            loadB(k0_c + KSTEP);
        }

        // ---- WMMA on current buffer ----
        const __bf16* Ab = Als[cur];
        const __bf16* Bb = Bls[cur];
#pragma unroll
        for (int s = 0; s < 2; ++s) {
            int ar = wave * 32 + s * 16 + ml;
            v8bf alo = *(const v8bf*)&Ab[ar * LDSS + hi * 8];
            v8bf ahi = *(const v8bf*)&Ab[ar * LDSS + 16 + hi * 8];
            v16bf afrag = __builtin_shufflevector(alo, ahi,
                0, 1, 2, 3, 4, 5, 6, 7, 8, 9, 10, 11, 12, 13, 14, 15);
#pragma unroll
            for (int j = 0; j < 4; ++j) {
                int br = j * 16 + ml;
                v8bf blo = *(const v8bf*)&Bb[br * LDSS + hi * 8];
                v8bf bhi = *(const v8bf*)&Bb[br * LDSS + 16 + hi * 8];
                v16bf bfrag = __builtin_shufflevector(blo, bhi,
                    0, 1, 2, 3, 4, 5, 6, 7, 8, 9, 10, 11, 12, 13, 14, 15);
                acc[s][j] = __builtin_amdgcn_wmma_f32_16x16x32_bf16(
                    false, afrag, false, bfrag, (short)0, acc[s][j], false, false);
            }
        }

        if (has_next) stage(cur ^ 1);         // wait loads + ds_store after WMMA
        __syncthreads();
        cur ^= 1;
        tap_c = tap_n; kc_c = kc_n; k0_c += KSTEP;
    }

    // ---- epilogue: D[lane][v] -> m = wave*32 + s*16 + hi*8 + v, n = j*16 + ml ----
#pragma unroll
    for (int s = 0; s < 2; ++s) {
#pragma unroll
        for (int j = 0; j < 4; ++j) {
            int n = nBase + j * 16 + ml;
            float sc = scale[n], tt = shift[n];
#pragma unroll
            for (int v = 0; v < 8; ++v) {
                int m = mBase + wave * 32 + s * 16 + hi * 8 + v;
                float o = acc[s][j][v] * sc + tt;
                if (do_silu) o = silu_fast(o);
                if (mode == 0) {
                    Cf[(size_t)m * ldC + n] = o;
                } else if (mode == 1) {
                    Cb[(size_t)m * ldC + n] = (__bf16)o;
                } else {
                    int in = m >> 12, sp = m & 4095;
                    size_t idx = ((size_t)in * 256 + n) * 4096 + sp;
                    outf[idx] = o + xres[idx] + y2res[(size_t)m * 256 + n];
                }
            }
        }
    }
}

// ---------------------------------------------------------------------------
// DCNv4 bilinear gather core: one wave per (pixel, group), lane = channel
//   value: [16384][256] f32 NHWC (G=8, Cg=32)
//   om:    [16384][256] f32 (per g: 18 offsets + 9 masks at g*27)
//   core:  [16384][256] bf16 out
// ---------------------------------------------------------------------------
__global__ __launch_bounds__(256) void dcn_core_k(const float* __restrict__ value,
                                                  const float* __restrict__ om,
                                                  __bf16* __restrict__ core) {
    int item  = blockIdx.x * 8 + (threadIdx.x >> 5);   // pixel*8 + g
    int c     = threadIdx.x & 31;
    int pixel = item >> 3;
    int g     = item & 7;
    int n = pixel >> 12, h = (pixel >> 6) & 63, w = pixel & 63;

    const float* ob = om + (size_t)pixel * 256 + g * 27;
    const float* vb = value + (((size_t)n) << 12) * 256 + g * 32 + c;

    float acc = 0.0f;
#pragma unroll
    for (int kp = 0; kp < 9; ++kp) {
        float offx = ob[2 * kp];
        float offy = ob[2 * kp + 1];
        float mk   = ob[18 + kp];
        float px = (float)w + (float)(kp % 3 - 1) + offx;
        float py = (float)h + (float)(kp / 3 - 1) + offy;
        float x0 = floorf(px), y0 = floorf(py);
        float lx = px - x0, ly = py - y0;
        int xi = (int)x0, yi = (int)y0;

        auto tap = [&](int y, int x, float wg) -> float {
            bool valid = (y >= 0) & (y < 64) & (x >= 0) & (x < 64);
            int yc = y < 0 ? 0 : (y > 63 ? 63 : y);
            int xc = x < 0 ? 0 : (x > 63 ? 63 : x);
            float vv = vb[((size_t)(yc << 6) + xc) * 256];
            return valid ? vv * wg : 0.0f;
        };
        float s = tap(yi, xi, (1.0f - ly) * (1.0f - lx))
                + tap(yi, xi + 1, (1.0f - ly) * lx)
                + tap(yi + 1, xi, ly * (1.0f - lx))
                + tap(yi + 1, xi + 1, ly * lx);
        acc += s * mk;
    }
    core[(size_t)pixel * 256 + g * 32 + c] = (__bf16)acc;
}

// ---------------------------------------------------------------------------
// Host launcher
// ---------------------------------------------------------------------------
extern "C" void kernel_launch(void* const* d_in, const int* in_sizes, int n_in,
                              void* d_out, int out_size, void* d_ws, size_t ws_size,
                              hipStream_t stream) {
    const float* x      = (const float*)d_in[0];
    const float* cv1_w  = (const float*)d_in[1];
    const float* cv1_g  = (const float*)d_in[2];
    const float* cv1_b  = (const float*)d_in[3];
    const float* cv1_m  = (const float*)d_in[4];
    const float* cv1_v  = (const float*)d_in[5];
    const float* cv2_w  = (const float*)d_in[6];
    const float* cv2_g  = (const float*)d_in[7];
    const float* cv2_b  = (const float*)d_in[8];
    const float* cv2_m  = (const float*)d_in[9];
    const float* cv2_v  = (const float*)d_in[10];
    const float* val_w  = (const float*)d_in[11];
    const float* val_b  = (const float*)d_in[12];
    const float* om_w   = (const float*)d_in[13];
    const float* om_b   = (const float*)d_in[14];
    const float* outp_w = (const float*)d_in[15];
    const float* outp_b = (const float*)d_in[16];
    const float* bn3_g  = (const float*)d_in[17];
    const float* bn3_b  = (const float*)d_in[18];
    const float* bn3_m  = (const float*)d_in[19];
    const float* bn3_v  = (const float*)d_in[20];
    const float* pw1_w  = (const float*)d_in[21];
    const float* pw1_b  = (const float*)d_in[22];
    const float* pw2_w  = (const float*)d_in[23];
    const float* pw2_b  = (const float*)d_in[24];
    float* out = (float*)d_out;

    const size_t M = 16384; // 4*64*64 pixels
    size_t off = 0;
    char* ws = (char*)d_ws;
    auto alloc = [&](size_t bytes) -> char* {
        char* p = ws + off;
        off = (off + bytes + 255) & ~(size_t)255;
        return p;
    };

    __bf16* x_bf   = (__bf16*)alloc(M * 256 * 2);
    __bf16* y1_bf  = (__bf16*)alloc(M * 128 * 2);
    float*  y2_f   = (float*) alloc(M * 256 * 4);
    float*  val_f  = (float*) alloc(M * 256 * 4);
    float*  om_f   = (float*) alloc(M * 256 * 4);
    __bf16* core_b = (__bf16*)alloc(M * 256 * 2);
    __bf16* d1_bf  = (__bf16*)alloc(M * 256 * 2);
    __bf16* d2_bf  = (__bf16*)alloc(M * 768 * 2);
    __bf16* wt_cv1 = (__bf16*)alloc((size_t)128 * 2304 * 2);
    __bf16* wt_cv2 = (__bf16*)alloc((size_t)256 * 1152 * 2);
    __bf16* wt_val = (__bf16*)alloc((size_t)256 * 256 * 2);
    __bf16* wt_om  = (__bf16*)alloc((size_t)256 * 256 * 2);
    __bf16* wt_out = (__bf16*)alloc((size_t)256 * 256 * 2);
    __bf16* wt_pw1 = (__bf16*)alloc((size_t)768 * 256 * 2);
    __bf16* wt_pw2 = (__bf16*)alloc((size_t)256 * 768 * 2);
    float* s_y1  = (float*)alloc(128 * 4); float* t_y1  = (float*)alloc(128 * 4);
    float* s_y2  = (float*)alloc(256 * 4); float* t_y2  = (float*)alloc(256 * 4);
    float* s_val = (float*)alloc(256 * 4); float* t_val = (float*)alloc(256 * 4);
    float* s_om  = (float*)alloc(256 * 4); float* t_om  = (float*)alloc(256 * 4);
    float* s_d1  = (float*)alloc(256 * 4); float* t_d1  = (float*)alloc(256 * 4);
    float* s_pw1 = (float*)alloc(768 * 4); float* t_pw1 = (float*)alloc(768 * 4);
    float* s_pw2 = (float*)alloc(256 * 4); float* t_pw2 = (float*)alloc(256 * 4);
    (void)ws_size; (void)n_in; (void)in_sizes; (void)out_size;

    // ---- prep ----
    pack_x_k<<<(M * 256) / 256, 256, 0, stream>>>(x, x_bf);
    prep_w_k<<<(128 * 2304 + 255) / 256, 256, 0, stream>>>(cv1_w,  wt_cv1, 128, 256, 9, 128);
    prep_w_k<<<(256 * 1152 + 255) / 256, 256, 0, stream>>>(cv2_w,  wt_cv2, 256, 128, 9, 256);
    prep_w_k<<<(256 * 256  + 255) / 256, 256, 0, stream>>>(val_w,  wt_val, 256, 256, 1, 256);
    prep_w_k<<<(256 * 256  + 255) / 256, 256, 0, stream>>>(om_w,   wt_om,  216, 256, 1, 256);
    prep_w_k<<<(256 * 256  + 255) / 256, 256, 0, stream>>>(outp_w, wt_out, 256, 256, 1, 256);
    prep_w_k<<<(768 * 256  + 255) / 256, 256, 0, stream>>>(pw1_w,  wt_pw1, 768, 256, 1, 768);
    prep_w_k<<<(256 * 768  + 255) / 256, 256, 0, stream>>>(pw2_w,  wt_pw2, 256, 768, 1, 256);
    bn_affine_k<<<1, 256, 0, stream>>>(cv1_g, cv1_b, cv1_m, cv1_v, s_y1, t_y1, 128);
    bn_affine_k<<<1, 256, 0, stream>>>(cv2_g, cv2_b, cv2_m, cv2_v, s_y2, t_y2, 256);
    bias_affine_k<<<1, 256, 0, stream>>>(val_b, s_val, t_val, 256, 256);
    bias_affine_k<<<1, 256, 0, stream>>>(om_b,  s_om,  t_om,  216, 256);
    outp_bn_affine_k<<<1, 256, 0, stream>>>(outp_b, bn3_g, bn3_b, bn3_m, bn3_v, s_d1, t_d1, 256);
    bias_affine_k<<<3, 256, 0, stream>>>(pw1_b, s_pw1, t_pw1, 768, 768);
    bias_affine_k<<<1, 256, 0, stream>>>(pw2_b, s_pw2, t_pw2, 256, 256);

    auto gemm = [&](const __bf16* Ap, const __bf16* Bp, const float* sp, const float* tp,
                    int K, int Cin, int N, int ldC, int ntaps, int silu, int mode,
                    float* Cf, __bf16* Cb, const float* xr, const float* yr, float* of) {
        dim3 grid(N / NBLK, M / MBLK);
        gemm_wmma_k<<<grid, 256, 0, stream>>>(Ap, Bp, sp, tp, K, Cin, N, ldC, ntaps,
                                              silu, mode, Cf, Cb, xr, yr, of);
    };

    // y1 = silu(bn1(conv3x3(x)))            -> bf16 [M][128]
    gemm(x_bf, wt_cv1, s_y1, t_y1, 2304, 256, 128, 128, 9, 1, 1,
         nullptr, y1_bf, nullptr, nullptr, nullptr);
    // y2 = silu(bn2(conv3x3(y1)))           -> f32 [M][256]
    gemm(y1_bf, wt_cv2, s_y2, t_y2, 1152, 128, 256, 256, 9, 1, 0,
         y2_f, nullptr, nullptr, nullptr, nullptr);
    // value = x @ val_w.T + val_b           -> f32 [M][256]
    gemm(x_bf, wt_val, s_val, t_val, 256, 256, 256, 256, 1, 0, 0,
         val_f, nullptr, nullptr, nullptr, nullptr);
    // om = x @ om_w.T + om_b                -> f32 [M][256] (216 valid, padded)
    gemm(x_bf, wt_om, s_om, t_om, 256, 256, 256, 256, 1, 0, 0,
         om_f, nullptr, nullptr, nullptr, nullptr);
    // deformable bilinear core              -> bf16 [M][256]
    dcn_core_k<<<M * 8 / 8, 256, 0, stream>>>(val_f, om_f, core_b);
    // d1 = bn3(core @ outp_w.T + outp_b)    -> bf16 [M][256]
    gemm(core_b, wt_out, s_d1, t_d1, 256, 256, 256, 256, 1, 0, 1,
         nullptr, d1_bf, nullptr, nullptr, nullptr);
    // d2 = silu(d1 @ pw1_w.T + pw1_b)       -> bf16 [M][768]
    gemm(d1_bf, wt_pw1, s_pw1, t_pw1, 256, 256, 768, 768, 1, 1, 1,
         nullptr, d2_bf, nullptr, nullptr, nullptr);
    // out = x + y2 + (d2 @ pw2_w.T + pw2_b) -> NCHW f32
    gemm(d2_bf, wt_pw2, s_pw2, t_pw2, 768, 768, 256, 256, 1, 0, 2,
         nullptr, nullptr, x, y2_f, out);
}
